// DLRM_56435870269512
// MI455X (gfx1250) — matrix operand order repaired and checked by reading, not compiled, not run on hardware
//
#include <hip/hip_runtime.h>

// ---------------------------------------------------------------------------
// DLRM forward on MI455X (gfx1250, wave32, WMMA).
// Precision: bf16 inputs to v_wmma_f32_16x16x32_bf16, f32 accumulate.
// GEMM staging uses GLOBAL_LOAD_ASYNC_TO_LDS_B128 (ASYNCcnt) with
// double-buffered LDS; one barrier per K-step.
// ---------------------------------------------------------------------------

#define Bsz 4096
#define Lq  10
#define Vv  50000
#define Dd  128
#define Tt  26

typedef __attribute__((ext_vector_type(16))) __bf16 v16bf;
typedef __attribute__((ext_vector_type(8)))  float  v8f;

__device__ __forceinline__ unsigned short f2bf(float f) {
    unsigned int u = __float_as_uint(f);
    unsigned int r = u + 0x7FFFu + ((u >> 16) & 1u);   // round-to-nearest-even
    return (unsigned short)(r >> 16);
}

// Build a 16-element bf16 fragment from two 16-byte chunks (8 bf16 each).
// Matches the CDNA5 16-bit A/B VGPR layouts.
__device__ __forceinline__ v16bf ldfrag2(const unsigned short* lo,
                                         const unsigned short* hi) {
    v16bf v;
    __builtin_memcpy(&v, lo, 16);
    __builtin_memcpy(reinterpret_cast<char*>(&v) + 16, hi, 16);
    return v;
}

// --- async global->LDS copy (16B per lane), tracked by ASYNCcnt ------------
__device__ __forceinline__ void async_copy16(unsigned short* lds,
                                             const unsigned short* g) {
    // generic pointer to __shared__: low 32 bits are the LDS byte address
    unsigned lds_addr = (unsigned)(unsigned long long)lds;
    asm volatile("global_load_async_to_lds_b128 %0, %1, off"
                 :: "v"(lds_addr), "v"(g)
                 : "memory");
}

__device__ __forceinline__ void wait_async0() {
#if __has_builtin(__builtin_amdgcn_s_wait_asynccnt)
    __builtin_amdgcn_s_wait_asynccnt(0);
#else
    asm volatile("s_wait_asynccnt 0x0" ::: "memory");
#endif
}

// ---------------------------------------------------------------------------
// Pack fp32 weight [N x K] -> bf16 [Npad x Kpad] row-major, zero padded.
// Also used for the numerical features (N=B rows, K=13 -> Kpad=32).
// ---------------------------------------------------------------------------
__global__ void pack_w_kernel(const float* __restrict__ src,
                              unsigned short* __restrict__ dst,
                              int N, int K, int Kpad, int total) {
    int i = blockIdx.x * blockDim.x + threadIdx.x;
    if (i >= total) return;
    int n = i / Kpad, k = i % Kpad;
    float v = (n < N && k < K) ? src[(size_t)n * K + k] : 0.0f;
    dst[i] = f2bf(v);
}

// ---------------------------------------------------------------------------
// Y = relu(X * W^T + bias).  X: [M x Kpad] bf16, W: [Npad x Kpad] bf16.
// Block = 128 threads (4 waves). Block tile 64(M) x 64(N), K-step 32.
// Double-buffered LDS staged via async global->LDS copies (ASYNCcnt).
// Wave w owns M-rows [16w,16w+16); computes 4 N-tiles -> 4 WMMAs per K-step.
// If Yf != nullptr, writes fp32 to Yf[m*Nreal + n] for n < Nreal instead.
// ---------------------------------------------------------------------------
__global__ __launch_bounds__(128)
void gemm_relu_wmma(const unsigned short* __restrict__ X,
                    const unsigned short* __restrict__ W,
                    const float* __restrict__ bias, int Nreal,
                    unsigned short* __restrict__ Y, int ldy,
                    float* __restrict__ Yf, int Kpad) {
    __shared__ unsigned short sX[2][64 * 32];
    __shared__ unsigned short sW[2][64 * 32];

    const int m0   = blockIdx.x * 64;
    const int n0   = blockIdx.y * 64;
    const int tid  = threadIdx.x;
    const int lane = tid & 31;
    const int wid  = tid >> 5;
    const int lrow = tid >> 1;          // 0..63 staging row
    const int lcol = (tid & 1) * 16;    // element 0 or 16 within K-step

    v8f acc[4] = {};

    const int m     = lane & 15;
    const int abase = (lane < 16) ? 0 : 8;    // A frag: K chunks {b..b+7, b+16..b+23}
    const int kb    = (lane < 16) ? 0 : 8;    // B frag: K chunks {b..b+7, b+16..b+23}

    auto stage = [&](int buf, int kk) {
        const unsigned short* gx = X + (size_t)(m0 + lrow) * Kpad + kk + lcol;
        const unsigned short* gw = W + (size_t)(n0 + lrow) * Kpad + kk + lcol;
        unsigned short* dx = &sX[buf][lrow * 32 + lcol];
        unsigned short* dw = &sW[buf][lrow * 32 + lcol];
        async_copy16(dx,     gx);
        async_copy16(dx + 8, gx + 8);
        async_copy16(dw,     gw);
        async_copy16(dw + 8, gw + 8);
    };

    int buf = 0;
    stage(0, 0);
    wait_async0();
    __syncthreads();

    for (int kk = 0; kk < Kpad; kk += 32) {
        const bool more = (kk + 32 < Kpad);
        if (more) stage(buf ^ 1, kk + 32);   // prefetch next tile into other buffer

        v16bf af = ldfrag2(&sX[buf][(wid * 16 + m) * 32 + abase],
                           &sX[buf][(wid * 16 + m) * 32 + abase + 16]);
#pragma unroll
        for (int j = 0; j < 4; ++j) {
            v16bf bfr = ldfrag2(&sW[buf][(j * 16 + m) * 32 + kb],
                                &sW[buf][(j * 16 + m) * 32 + kb + 16]);
            acc[j] = __builtin_amdgcn_wmma_f32_16x16x32_bf16(
                false, af, false, bfr, (short)0, acc[j], false, false);
        }

        if (more) {
            wait_async0();       // next tile resident in LDS
            __syncthreads();     // all waves done reading current buffer
            buf ^= 1;
        }
    }

    // Epilogue: C layout = VGPR v -> row v (+8 for lanes 16..31), col = lane&15.
    const int col  = lane & 15;
    const int rofs = (lane >= 16) ? 8 : 0;
#pragma unroll
    for (int j = 0; j < 4; ++j) {
        int n_g  = n0 + j * 16 + col;
        float bv = (n_g < Nreal) ? bias[n_g] : 0.0f;
#pragma unroll
        for (int v = 0; v < 8; ++v) {
            int   m_g = m0 + wid * 16 + v + rofs;
            float r   = fmaxf(acc[j][v] + bv, 0.0f);
            if (Yf) {
                if (n_g < Nreal) Yf[(size_t)m_g * Nreal + n_g] = r;
            } else {
                Y[(size_t)m_g * ldy + n_g] = f2bf(r);
            }
        }
    }
}

// ---------------------------------------------------------------------------
// z[b] (32 x 128 bf16): row 0 = bottom-MLP output (bf16 copy),
// rows 1..26 = sum_l emb_tables[t][idx[t][b][l]], rows 27..31 = 0.
// One wave per (b, row); lane covers 4 of the 128 dims (float4 row loads).
// ---------------------------------------------------------------------------
__global__ void build_z_kernel(const int* __restrict__ idx,
                               const float* __restrict__ tbl,
                               const unsigned short* __restrict__ bot,
                               unsigned short* __restrict__ z) {
    const int b    = blockIdx.x;
    const int r    = blockIdx.y * blockDim.y + threadIdx.y;  // 0..31
    const int lane = threadIdx.x;
    unsigned short* dst = z + ((size_t)b * 32 + r) * Dd + lane * 4;
    unsigned int*   d   = reinterpret_cast<unsigned int*>(dst);
    if (r == 0) {
        const unsigned int* s = reinterpret_cast<const unsigned int*>(
            bot + (size_t)b * Dd + lane * 4);
        d[0] = s[0]; d[1] = s[1];
    } else if (r <= Tt) {
        const int  t  = r - 1;
        const int* id = idx + ((size_t)t * Bsz + b) * Lq;
        float4 acc = make_float4(0.f, 0.f, 0.f, 0.f);
        for (int l = 0; l < Lq; ++l) {
            const float4* row = reinterpret_cast<const float4*>(
                tbl + ((size_t)t * Vv + id[l]) * Dd);
            if (l + 1 < Lq)
                __builtin_prefetch(tbl + ((size_t)t * Vv + id[l + 1]) * Dd + lane * 4, 0, 1);
            float4 v = row[lane];
            acc.x += v.x; acc.y += v.y; acc.z += v.z; acc.w += v.w;
        }
        d[0] = (unsigned)f2bf(acc.x) | ((unsigned)f2bf(acc.y) << 16);
        d[1] = (unsigned)f2bf(acc.z) | ((unsigned)f2bf(acc.w) << 16);
    } else {
        d[0] = 0u; d[1] = 0u;
    }
}

// ---------------------------------------------------------------------------
// Interaction: per sample, Z = z z^T (32x32 padded; 2x2 WMMA tiles, K=128).
// Writes top_in[b] (512 bf16): [0..127] = bot, [128..478] = strict lower
// triangle of the 27x27 block (np.tril_indices order), [479..511] = 0.
// One wave per sample, 8 waves per block.
// ---------------------------------------------------------------------------
__device__ __forceinline__ void tril_scatter(unsigned short* o, int i, int j,
                                             float val) {
    if (i <= Tt && j < i)  // i in 0..26, strictly lower triangle
        o[128 + (i * (i - 1)) / 2 + j] = f2bf(val);
}

__global__ __launch_bounds__(256)
void interact_wmma(const unsigned short* __restrict__ z,
                   unsigned short* __restrict__ tin) {
    const int b    = blockIdx.x * blockDim.y + threadIdx.y;
    const int lane = threadIdx.x;
    const unsigned short* zb = z + (size_t)b * 32 * Dd;

    v8f a00 = {}, a01 = {}, a10 = {}, a11 = {};
    const int m     = lane & 15;
    const int abase = (lane < 16) ? 0 : 8;
    const int kb    = (lane < 16) ? 0 : 8;

#pragma unroll
    for (int kk = 0; kk < Dd; kk += 32) {
        v16bf af0 = ldfrag2(zb + (size_t)(m)*Dd + kk + abase,
                            zb + (size_t)(m)*Dd + kk + abase + 16);
        v16bf af1 = ldfrag2(zb + (size_t)(16 + m) * Dd + kk + abase,
                            zb + (size_t)(16 + m) * Dd + kk + abase + 16);
        v16bf bf0 = ldfrag2(zb + (size_t)(m)*Dd + kk + kb,
                            zb + (size_t)(m)*Dd + kk + kb + 16);
        v16bf bf1 = ldfrag2(zb + (size_t)(16 + m) * Dd + kk + kb,
                            zb + (size_t)(16 + m) * Dd + kk + kb + 16);
        a00 = __builtin_amdgcn_wmma_f32_16x16x32_bf16(false, af0, false, bf0,
                                                      (short)0, a00, false, false);
        a01 = __builtin_amdgcn_wmma_f32_16x16x32_bf16(false, af0, false, bf1,
                                                      (short)0, a01, false, false);
        a10 = __builtin_amdgcn_wmma_f32_16x16x32_bf16(false, af1, false, bf0,
                                                      (short)0, a10, false, false);
        a11 = __builtin_amdgcn_wmma_f32_16x16x32_bf16(false, af1, false, bf1,
                                                      (short)0, a11, false, false);
    }

    unsigned short* o = tin + (size_t)b * 512;
    // bot features = z row 0
    reinterpret_cast<uint2*>(o)[lane] = reinterpret_cast<const uint2*>(zb)[lane];
    // zero pad tail [479,512)
    for (int i = 479 + lane; i < 512; i += 32) o[i] = 0;
    // lower-triangle scatter
    const int col  = lane & 15;
    const int rofs = (lane >= 16) ? 8 : 0;
#pragma unroll
    for (int v = 0; v < 8; ++v) {
        int i0 = v + rofs;
        tril_scatter(o, i0,      col,      a00[v]);
        tril_scatter(o, i0,      16 + col, a01[v]);
        tril_scatter(o, 16 + i0, col,      a10[v]);
        tril_scatter(o, 16 + i0, 16 + col, a11[v]);
    }
}

// ---------------------------------------------------------------------------
// Host-side orchestration.
// ---------------------------------------------------------------------------
extern "C" void kernel_launch(void* const* d_in, const int* in_sizes, int n_in,
                              void* d_out, int out_size, void* d_ws, size_t ws_size,
                              hipStream_t stream) {
    (void)in_sizes; (void)n_in; (void)out_size; (void)ws_size;

    const float* numerical = (const float*)d_in[0];
    const int*   indices   = (const int*)d_in[1];
    const float* tables    = (const float*)d_in[2];
    const float* bw[3]     = {(const float*)d_in[3], (const float*)d_in[4],
                              (const float*)d_in[5]};
    const float* bb[3]     = {(const float*)d_in[6], (const float*)d_in[7],
                              (const float*)d_in[8]};
    const float* tw[5]     = {(const float*)d_in[9], (const float*)d_in[10],
                              (const float*)d_in[11], (const float*)d_in[12],
                              (const float*)d_in[13]};
    const float* tb[5]     = {(const float*)d_in[14], (const float*)d_in[15],
                              (const float*)d_in[16], (const float*)d_in[17],
                              (const float*)d_in[18]};

    size_t off = 0;  // in bf16 elements
    auto alloc = [&](size_t elems) -> unsigned short* {
        unsigned short* p = (unsigned short*)d_ws + off;
        off += (elems + 127) & ~(size_t)127;   // 256B alignment
        return p;
    };

    // packed bf16 weights [Npad x Kpad]
    unsigned short* pbw1 = alloc(512 * 32);
    unsigned short* pbw2 = alloc(256 * 512);
    unsigned short* pbw3 = alloc(128 * 256);
    unsigned short* ptw1 = alloc((size_t)1024 * 512);
    unsigned short* ptw2 = alloc((size_t)1024 * 1024);
    unsigned short* ptw3 = alloc((size_t)512 * 1024);
    unsigned short* ptw4 = alloc(256 * 512);
    unsigned short* ptw5 = alloc(64 * 256);      // N=1 padded to 64
    // activations (bf16)
    unsigned short* x0  = alloc((size_t)Bsz * 32);        // numerical padded
    unsigned short* a1  = alloc((size_t)Bsz * 512);
    unsigned short* a2  = alloc((size_t)Bsz * 256);
    unsigned short* bot = alloc((size_t)Bsz * 128);
    unsigned short* zb  = alloc((size_t)Bsz * 32 * Dd);   // padded z
    unsigned short* tin = alloc((size_t)Bsz * 512);       // top_in padded 479->512
    unsigned short* t2  = alloc((size_t)Bsz * 1024);
    // aliased (lifetimes disjoint)
    unsigned short* t1  = zb;   // 8 MB needed, zb region is 32 MB
    unsigned short* t3  = a1;   // 4 MB needed, a1 region is 4 MB
    unsigned short* t4  = a2;   // 2 MB needed, a2 region is 2 MB

    auto pack = [&](const float* s, unsigned short* dpk, int N, int K, int Kp,
                    int Np) {
        int tot = Np * Kp;
        pack_w_kernel<<<dim3((tot + 255) / 256), dim3(256), 0, stream>>>(
            s, dpk, N, K, Kp, tot);
    };
    pack(bw[0], pbw1, 512, 13, 32, 512);
    pack(bw[1], pbw2, 256, 512, 512, 256);
    pack(bw[2], pbw3, 128, 256, 256, 128);
    pack(tw[0], ptw1, 1024, 479, 512, 1024);
    pack(tw[1], ptw2, 1024, 1024, 1024, 1024);
    pack(tw[2], ptw3, 512, 1024, 1024, 512);
    pack(tw[3], ptw4, 256, 512, 512, 256);
    pack(tw[4], ptw5, 1, 256, 256, 64);
    pack(numerical, x0, Bsz, 13, 32, Bsz);

    auto gemm = [&](const unsigned short* X, const unsigned short* W,
                    const float* bias, int Nreal, int Npad, int Kpad,
                    unsigned short* Y, float* Yf) {
        gemm_relu_wmma<<<dim3(Bsz / 64, Npad / 64), dim3(128), 0, stream>>>(
            X, W, bias, Nreal, Y, Npad, Yf, Kpad);
    };

    // bottom MLP: 13 -> 512 -> 256 -> 128
    gemm(x0, pbw1, bb[0], 512, 512, 32, a1, nullptr);
    gemm(a1, pbw2, bb[1], 256, 256, 512, a2, nullptr);
    gemm(a2, pbw3, bb[2], 128, 128, 256, bot, nullptr);

    // embedding pooling + z assembly
    build_z_kernel<<<dim3(Bsz, 4), dim3(32, 8), 0, stream>>>(indices, tables,
                                                             bot, zb);
    // pairwise interaction + top_in assembly
    interact_wmma<<<dim3(Bsz / 8), dim3(32, 8), 0, stream>>>(zb, tin);

    // top MLP: 479 -> 1024 -> 1024 -> 512 -> 256 -> 1
    gemm(tin, ptw1, tb[0], 1024, 1024, 512, t1, nullptr);
    gemm(t1,  ptw2, tb[1], 1024, 1024, 1024, t2, nullptr);
    gemm(t2,  ptw3, tb[2], 512, 512, 1024, t3, nullptr);
    gemm(t3,  ptw4, tb[3], 256, 256, 512, t4, nullptr);
    gemm(t4,  ptw5, tb[4], 1, 64, 256, nullptr, (float*)d_out);
}
